// SimpleModel_13357348291070
// MI455X (gfx1250) — matrix operand order repaired and checked by reading, not compiled, not run on hardware
//
#include <hip/hip_runtime.h>

#define H 128

typedef __attribute__((ext_vector_type(16))) __bf16 v16bf;
typedef __attribute__((ext_vector_type(8)))  float  v8f;

// ---------- fp32 <-> bf16 split helpers (round-to-nearest-even) ----------
__device__ inline unsigned short f32_to_bf16_rne(float f) {
    unsigned u = __builtin_bit_cast(unsigned, f);
    unsigned r = u + 0x7FFFu + ((u >> 16) & 1u);
    return (unsigned short)(r >> 16);
}
__device__ inline float bf16_bits_to_f32(unsigned short b) {
    unsigned u = ((unsigned)b) << 16;
    return __builtin_bit_cast(float, u);
}
__device__ inline __bf16 bits_to_bf16(unsigned short b) {
    return __builtin_bit_cast(__bf16, b);
}

// ---------------------------------------------------------------------------
// GEMM: Out[n, j] = sum_k A[n,k] * W[j,k]  (+ bias[j])
// bf16x3 split-precision on v_wmma_f32_16x16x32_bf16 (~fp32 accuracy).
// Block = 256 threads (8 waves); each wave computes a 16-row strip.
// W (hi/lo bf16 planes) staged in LDS: 64 KB.
// ---------------------------------------------------------------------------
__global__ __launch_bounds__(256)
void gemm_xWt_bf16x3(const float* __restrict__ A, const float* __restrict__ W,
                     const float* __restrict__ bias, float* __restrict__ Out,
                     int nRows) {
    __shared__ unsigned short sWhi[H][H];
    __shared__ unsigned short sWlo[H][H];

    const int tid = threadIdx.x;
    for (int idx = tid; idx < H * H; idx += 256) {
        float w = W[idx];
        unsigned short hb = f32_to_bf16_rne(w);
        unsigned short lb = f32_to_bf16_rne(w - bf16_bits_to_f32(hb));
        sWhi[idx >> 7][idx & (H - 1)] = hb;
        sWlo[idx >> 7][idx & (H - 1)] = lb;
    }
    __syncthreads();

    const int wave  = tid >> 5;
    const int lane  = tid & 31;
    const int khalf = lane >> 4;   // which K-half this lane holds
    const int mn    = lane & 15;   // M for A-frag, N for B/C/D frags
    const int r0    = (blockIdx.x * 8 + wave) * 16;
    if (r0 >= nRows) return;       // wave-uniform exit (EXEC stays all-ones for WMMA)

    // ---- load A fragments for all K once (hi/lo split), 16 rows per wave ----
    // A 16x32 bf16 layout: VGPR v holds packed K pair {base + 2*(v&3), +1} + khalf*8,
    // base = 16 for v>=4 -> pairs are contiguous in memory: use float2 loads.
    int row = r0 + mn; if (row >= nRows) row = nRows - 1;   // clamp (dup rows, stores guarded)
    const float* __restrict__ arow = A + (size_t)row * H;
    v16bf aHi[4], aLo[4];
#pragma unroll
    for (int kt = 0; kt < 4; ++kt) {
#pragma unroll
        for (int v = 0; v < 8; ++v) {
            int base = (v < 4) ? 0 : 16;
            int K = kt * 32 + base + (v & 3) * 2 + khalf * 8;
            float2 x2 = *(const float2*)(arow + K);
            unsigned short hb0 = f32_to_bf16_rne(x2.x);
            unsigned short lb0 = f32_to_bf16_rne(x2.x - bf16_bits_to_f32(hb0));
            unsigned short hb1 = f32_to_bf16_rne(x2.y);
            unsigned short lb1 = f32_to_bf16_rne(x2.y - bf16_bits_to_f32(hb1));
            aHi[kt][2 * v]     = bits_to_bf16(hb0);
            aHi[kt][2 * v + 1] = bits_to_bf16(hb1);
            aLo[kt][2 * v]     = bits_to_bf16(lb0);
            aLo[kt][2 * v + 1] = bits_to_bf16(lb1);
        }
    }

    // ---- preload bias values (hoisted out of the hot loop) ----
    float bvals[8];
#pragma unroll
    for (int jt = 0; jt < 8; ++jt)
        bvals[jt] = bias ? bias[jt * 16 + mn] : 0.0f;

    const bool fullStrip = (r0 + 16 <= nRows);   // wave-uniform

    // ---- 8 output tiles of 16x16 along j ----
#pragma unroll
    for (int jt = 0; jt < 8; ++jt) {
        const int j0 = jt * 16;
        v8f acc = {};
#pragma unroll
        for (int kt = 0; kt < 4; ++kt) {
            v16bf bHi, bLo;
#pragma unroll
            for (int e = 0; e < 16; ++e) {
                int K = kt * 32 + khalf * 8 + (e & 7) + (e >> 3) * 16;      // B 32x16: same K order as A frag
                bHi[e] = bits_to_bf16(sWhi[j0 + mn][K]);  // B[K][n] = W[j0+n][K]
                bLo[e] = bits_to_bf16(sWlo[j0 + mn][K]);
            }
            acc = __builtin_amdgcn_wmma_f32_16x16x32_bf16(false, aHi[kt], false, bHi,
                                                          (short)0, acc, false, false);
            acc = __builtin_amdgcn_wmma_f32_16x16x32_bf16(false, aLo[kt], false, bHi,
                                                          (short)0, acc, false, false);
            acc = __builtin_amdgcn_wmma_f32_16x16x32_bf16(false, aHi[kt], false, bLo,
                                                          (short)0, acc, false, false);
        }
        const float bval = bvals[jt];
        if (fullStrip) {
#pragma unroll
            for (int v = 0; v < 8; ++v) {
                int orow = r0 + v + khalf * 8;        // D layout: M = vgpr + 8*(lane>=16)
                Out[(size_t)orow * H + j0 + mn] = acc[v] + bval;
            }
        } else {
#pragma unroll
            for (int v = 0; v < 8; ++v) {
                int orow = r0 + v + khalf * 8;
                if (orow < nRows) Out[(size_t)orow * H + j0 + mn] = acc[v] + bval;
            }
        }
    }
}

// ---------------------------------------------------------------------------
// Graph utility kernels
// ---------------------------------------------------------------------------
__global__ void init_kernel(float* deg_out, float* deg, float* out8, int n) {
    int i = blockIdx.x * blockDim.x + threadIdx.x;
    if (i < n) { deg_out[i] = 0.0f; deg[i] = 1.0f; }   // deg starts at 1 (self loop)
    if (i < 8) out8[i] = 0.0f;
}

__global__ void degree_kernel(const int* __restrict__ src, const int* __restrict__ dst,
                              float* deg_out, float* deg, int E) {
    int e = blockIdx.x * blockDim.x + threadIdx.x;
    if (e < E) {
        atomicAdd(&deg_out[src[e]], 1.0f);
        atomicAdd(&deg[dst[e]], 1.0f);
    }
}

__global__ void dinv_kernel(const float* __restrict__ deg, float* dinv, int n) {
    int i = blockIdx.x * blockDim.x + threadIdx.x;
    if (i < n) dinv[i] = rsqrtf(deg[i]);
}

__global__ void zero_kernel(float* p, long long n) {
    long long i = (long long)blockIdx.x * blockDim.x + threadIdx.x;
    long long stride = (long long)gridDim.x * blockDim.x;
    for (; i < n; i += stride) p[i] = 0.0f;
}

// one wave32 per edge: hnext[dst] += dinv[s]*dinv[d] * hw[src]
__global__ __launch_bounds__(256)
void scatter_kernel(const int* __restrict__ src, const int* __restrict__ dst,
                    const float* __restrict__ dinv, const float* __restrict__ hw,
                    float* __restrict__ hnext, int E) {
    int e = blockIdx.x * 8 + (threadIdx.x >> 5);
    if (e >= E) return;
    int lane = threadIdx.x & 31;
    int s = src[e], d = dst[e];
    float nrm = dinv[s] * dinv[d];
    const float* __restrict__ hs = hw + (size_t)s * H;
    float* __restrict__ hd = hnext + (size_t)d * H;
#pragma unroll
    for (int c = lane; c < H; c += 32)
        atomicAdd(&hd[c], nrm * hs[c]);
}

// h = relu(hnext + dinv[i]^2 * hw)   and accumulate sum(h*h) -> norm_acc
__global__ __launch_bounds__(256)
void finalize_kernel(const float* __restrict__ hnext, const float* __restrict__ hw,
                     const float* __restrict__ dinv, float* __restrict__ h,
                     float* norm_acc, int n) {
    long long idx = (long long)blockIdx.x * 256 + threadIdx.x;
    float sq = 0.0f;
    if (idx < (long long)n * H) {
        int i = (int)(idx >> 7);
        float sl = dinv[i]; sl *= sl;
        float t = hnext[idx] + sl * hw[idx];
        float v = t > 0.0f ? t : 0.0f;
        h[idx] = v;
        sq = v * v;
    }
#pragma unroll
    for (int o = 16; o > 0; o >>= 1) sq += __shfl_xor(sq, o, 32);
    __shared__ float wpart[8];
    if ((threadIdx.x & 31) == 0) wpart[threadIdx.x >> 5] = sq;
    __syncthreads();
    if (threadIdx.x < 8) {
        float t = wpart[threadIdx.x];
#pragma unroll
        for (int o = 4; o > 0; o >>= 1) t += __shfl_xor(t, o, 8);
        if (threadIdx.x == 0) atomicAdd(norm_acc, t);
    }
}

// one wave32 per edge: energy += 0.5 * (1/deg_out[s]) * ||h[s]-h[d]||^2
__global__ __launch_bounds__(256)
void energy_kernel(const int* __restrict__ src, const int* __restrict__ dst,
                   const float* __restrict__ deg_out, const float* __restrict__ h,
                   float* energy_acc, int E) {
    int wv = threadIdx.x >> 5;
    int e = blockIdx.x * 8 + wv;
    int lane = threadIdx.x & 31;
    float part = 0.0f;
    if (e < E) {
        int s = src[e], d = dst[e];
        const float* __restrict__ hs = h + (size_t)s * H;
        const float* __restrict__ hd = h + (size_t)d * H;
        float acc = 0.0f;
#pragma unroll
        for (int c = lane; c < H; c += 32) {
            float df = hs[c] - hd[c];
            acc += df * df;
        }
        part = acc * (0.5f / deg_out[s]);
    }
#pragma unroll
    for (int o = 16; o > 0; o >>= 1) part += __shfl_xor(part, o, 32);
    __shared__ float wpart[8];
    if (lane == 0) wpart[wv] = part;
    __syncthreads();
    if (threadIdx.x < 8) {
        float t = wpart[threadIdx.x];
#pragma unroll
        for (int o = 4; o > 0; o >>= 1) t += __shfl_xor(t, o, 8);
        if (threadIdx.x == 0) atomicAdd(energy_acc, t);
    }
}

// ---------------------------------------------------------------------------
extern "C" void kernel_launch(void* const* d_in, const int* in_sizes, int n_in,
                              void* d_out, int out_size, void* d_ws, size_t ws_size,
                              hipStream_t stream) {
    const float* x       = (const float*)d_in[0];
    const int*   ei      = (const int*)  d_in[1];
    const float* enc_w   = (const float*)d_in[2];
    const float* enc_b   = (const float*)d_in[3];
    const float* conv_ws = (const float*)d_in[4];
    float* out = (float*)d_out;   // out[0..L-1]=energies, out[L..2L-1]=norms

    const int N = in_sizes[0] / H;
    const int E = in_sizes[1] / 2;
    const int L = in_sizes[4] / (H * H);
    const int* src = ei;
    const int* dst = ei + E;

    float* ws       = (float*)d_ws;
    float* deg_out  = ws;  ws += N;
    float* deg      = ws;  ws += N;
    float* dinv     = ws;  ws += N;
    float* h        = ws;  ws += (size_t)N * H;
    float* hw       = ws;  ws += (size_t)N * H;
    float* hnext    = ws;  /* ws += (size_t)N * H; */

    const int tN    = (N + 255) / 256;
    const int tE    = (E + 255) / 256;
    const int wE    = (E + 7) / 8;          // one wave per edge
    const int gemmB = (N + 127) / 128;      // 128 rows per block

    init_kernel<<<tN, 256, 0, stream>>>(deg_out, deg, out, N);
    degree_kernel<<<tE, 256, 0, stream>>>(src, dst, deg_out, deg, E);
    dinv_kernel<<<tN, 256, 0, stream>>>(deg, dinv, N);

    // encoder: h = x @ enc_w.T + enc_b
    gemm_xWt_bf16x3<<<gemmB, 256, 0, stream>>>(x, enc_w, enc_b, h, N);

    const long long nh = (long long)N * H;
    for (int l = 0; l < L; ++l) {
        gemm_xWt_bf16x3<<<gemmB, 256, 0, stream>>>(h, conv_ws + (size_t)l * H * H,
                                                   nullptr, hw, N);
        zero_kernel<<<2048, 256, 0, stream>>>(hnext, nh);
        scatter_kernel<<<wE, 256, 0, stream>>>(src, dst, dinv, hw, hnext, E);
        finalize_kernel<<<(int)((nh + 255) / 256), 256, 0, stream>>>(
            hnext, hw, dinv, h, out + L + l, N);
        energy_kernel<<<wE, 256, 0, stream>>>(src, dst, deg_out, h, out + l, E);
    }
}
